// Model_15453292331282
// MI455X (gfx1250) — matrix operand order repaired and checked by reading, not compiled, not run on hardware
//
#include <hip/hip_runtime.h>
#include <hip/hip_bf16.h>

#define N_NODES 100000
#define N_EDGES 600000
#define N_EL    200000

typedef _Float16 f16_t;
typedef __attribute__((ext_vector_type(16))) _Float16 v16h;
typedef __attribute__((ext_vector_type(8)))  float    v8f;

// ---------- helpers ----------
__device__ __forceinline__ unsigned fflip(float f) {
    unsigned u = __float_as_uint(f);
    return u ^ (unsigned)(((int)u >> 31) | 0x80000000);
}
__device__ __forceinline__ float funflip(unsigned c) {
    unsigned v = c ^ (((int)c >= 0) ? 0xFFFFFFFFu : 0x80000000u);
    return __uint_as_float(v);
}
// CDNA5 wave32 WMMA f16 B-fragment K index for half i (ISA 7.12.2)
__device__ __forceinline__ int bfrag_k(int lane, int i) {
    return ((lane & 16) ? 8 : 0) + (i < 8 ? i : i + 8);
}
// Pack 8 consecutive f32 (a[0..7], optionally +b[0..7]) into frag halves [g*8..g*8+7]
__device__ __forceinline__ void cvt8(v16h& frag, int g, const float* a) {
    float4 a0 = *(const float4*)a, a1 = *(const float4*)(a + 4);
    frag[g * 8 + 0] = (f16_t)a0.x; frag[g * 8 + 1] = (f16_t)a0.y;
    frag[g * 8 + 2] = (f16_t)a0.z; frag[g * 8 + 3] = (f16_t)a0.w;
    frag[g * 8 + 4] = (f16_t)a1.x; frag[g * 8 + 5] = (f16_t)a1.y;
    frag[g * 8 + 6] = (f16_t)a1.z; frag[g * 8 + 7] = (f16_t)a1.w;
}
__device__ __forceinline__ void cvt8_add(v16h& frag, int g, const float* a, const float* b) {
    float4 a0 = *(const float4*)a, a1 = *(const float4*)(a + 4);
    float4 b0 = *(const float4*)b, b1 = *(const float4*)(b + 4);
    frag[g * 8 + 0] = (f16_t)(a0.x + b0.x); frag[g * 8 + 1] = (f16_t)(a0.y + b0.y);
    frag[g * 8 + 2] = (f16_t)(a0.z + b0.z); frag[g * 8 + 3] = (f16_t)(a0.w + b0.w);
    frag[g * 8 + 4] = (f16_t)(a1.x + b1.x); frag[g * 8 + 5] = (f16_t)(a1.y + b1.y);
    frag[g * 8 + 6] = (f16_t)(a1.z + b1.z); frag[g * 8 + 7] = (f16_t)(a1.w + b1.w);
}
__device__ __forceinline__ void zero8(v16h& frag, int g) {
    for (int i = 0; i < 8; i++) frag[g * 8 + i] = (f16_t)0.f;
}

// ---------- init ----------
__global__ void init_layer(float* agg, long na, unsigned* mx, float* den, int n4) {
    long i = (long)blockIdx.x * blockDim.x + threadIdx.x;
    long stride = (long)gridDim.x * blockDim.x;
    for (long k = i; k < na; k += stride) agg[k] = 0.f;
    unsigned neg = fflip(-3.0e38f);
    for (long k = i; k < n4; k += stride) { mx[k] = neg; den[k] = 0.f; }
}

// ---------- pack weights [IN,OUT] f32 -> WMMA B-fragment order f16 ----------
// layout: out[(((kk*(OUT/16))+ct)*32 + lane)*16 + i], k = kk*32 + bfrag_k(lane,i)
template <int IN, int OUT>
__global__ void pack_bfrag(const float* __restrict__ W, f16_t* __restrict__ out) {
    constexpr int KS = (IN + 31) / 32;
    constexpr int TOT = KS * (OUT / 16) * 32 * 16;
    int idx = blockIdx.x * blockDim.x + threadIdx.x;
    if (idx >= TOT) return;
    int i = idx & 15;
    int lane = (idx >> 4) & 31;
    int rest = idx >> 9;
    int ct = rest % (OUT / 16);
    int kk = rest / (OUT / 16);
    int k = kk * 32 + bfrag_k(lane, i);
    int col = ct * 16 + (lane & 15);
    out[idx] = (k < IN) ? (f16_t)W[k * OUT + col] : (f16_t)0.f;
}

// wea[j,h] = sum_k We[j, h*d+k]*ae[h,k] ; eta[t,h] = sum_k et[t, h*d+k]*ae[h,k]
__global__ void prep_edge_vecs(const float* __restrict__ We, const float* __restrict__ et,
                               const float* __restrict__ ae, float* wea, float* eta, int d) {
    int tid = blockIdx.x * blockDim.x + threadIdx.x;
    int OUT = 4 * d;
    if (tid < 16 * 4) {
        int j = tid >> 2, hh = tid & 3;
        float s = 0.f;
        for (int k = 0; k < d; k++) s += We[j * OUT + hh * d + k] * ae[hh * d + k];
        wea[tid] = s;
    }
    int t2 = tid - 64;
    if (t2 >= 0 && t2 < 22 * 4) {
        int t = t2 >> 2, hh = t2 & 3;
        float s = 0.f;
        for (int k = 0; k < d; k++) s += et[t * OUT + hh * d + k] * ae[hh * d + k];
        eta[t2] = s;
    }
}

// ---------- node projection: h = (x + nt[type]) @ W,  [N,IN]@[IN,OUT] via WMMA ----------
template <int IN, int OUT>
__global__ __launch_bounds__(256) void node_proj(const float* __restrict__ xin,
                                                 const int* __restrict__ ntype,
                                                 const float* __restrict__ ntab,
                                                 const f16_t* __restrict__ Wp,
                                                 float* __restrict__ hout, int nNodes) {
    int lane = threadIdx.x & 31, wid = threadIdx.x >> 5;
    int tile = blockIdx.x * (blockDim.x >> 5) + wid;
    int nb = tile * 16;
    if (nb >= nNodes) return;
    int row = nb + (lane & 15);
    int t = ntype[row];
    int base = (lane & 16) ? 8 : 0;
    constexpr int KS = (IN + 31) / 32;
    v16h a[KS];
    for (int kk = 0; kk < KS; kk++)
        for (int g = 0; g < 2; g++) {
            int k0 = kk * 32 + g * 16 + base;
            if (k0 < IN) cvt8_add(a[kk], g, xin + (long)row * IN + k0, ntab + t * IN + k0);
            else         zero8(a[kk], g);
        }
    const v16h* Bp = (const v16h*)Wp;
    int mh = (lane & 16) ? 8 : 0;
    for (int ct = 0; ct < OUT / 16; ct++) {
        int col = ct * 16 + (lane & 15);
        v8f acc = {};
        for (int kk = 0; kk < KS; kk++) {
            v16h b = Bp[(kk * (OUT / 16) + ct) * 32 + lane];
            acc = __builtin_amdgcn_wmma_f32_16x16x32_f16(false, a[kk], false, b,
                                                         (short)0, acc, false, false);
        }
        for (int r = 0; r < 8; r++)
            hout[(long)(nb + mh + r) * OUT + col] = acc[r];
    }
}

// ---------- per-node attention dots ----------
__global__ void score_dots(const float* __restrict__ h, const float* __restrict__ as_,
                           const float* __restrict__ ad_, float* ssrc, float* sdst,
                           int n, int d) {
    int idx = blockIdx.x * blockDim.x + threadIdx.x;
    if (idx >= n * 4) return;
    int node = idx >> 2, hh = idx & 3;
    const float* hp = h + (long)node * (4 * d) + hh * d;
    float s = 0.f, dd = 0.f;
    for (int k = 0; k < d; k++) { s += hp[k] * as_[hh * d + k]; dd += hp[k] * ad_[hh * d + k]; }
    ssrc[idx] = s; sdst[idx] = dd;
}

// ---------- edge score + segment max ----------
__global__ void edge_score(const float* __restrict__ ea, const int* __restrict__ eidx,
                           const int* __restrict__ etype, const float* __restrict__ ssrc,
                           const float* __restrict__ sdst, const float* __restrict__ wea,
                           const float* __restrict__ eta, float* __restrict__ sc,
                           unsigned* __restrict__ mx) {
    int e = blockIdx.x * blockDim.x + threadIdx.x;
    if (e >= N_EDGES) return;
    int s = eidx[e], dn = eidx[N_EDGES + e], t = etype[e];
    float av[16];
    *(float4*)(av + 0)  = *(const float4*)(ea + (long)e * 16 + 0);
    *(float4*)(av + 4)  = *(const float4*)(ea + (long)e * 16 + 4);
    *(float4*)(av + 8)  = *(const float4*)(ea + (long)e * 16 + 8);
    *(float4*)(av + 12) = *(const float4*)(ea + (long)e * 16 + 12);
    float4 s4 = *(const float4*)(ssrc + (long)s * 4);
    float4 d4 = *(const float4*)(sdst + (long)dn * 4);
    float4 t4 = *(const float4*)(eta + t * 4);
    float vv[4] = {s4.x + d4.x + t4.x, s4.y + d4.y + t4.y,
                   s4.z + d4.z + t4.z, s4.w + d4.w + t4.w};
    for (int hh = 0; hh < 4; hh++) {
        float v = vv[hh];
        for (int j = 0; j < 16; j++) v += av[j] * wea[j * 4 + hh];
        v = v > 0.f ? v : 0.2f * v;      // leaky_relu(0.2)
        vv[hh] = v;
        atomicMax(&mx[dn * 4 + hh], fflip(v));
    }
    *(float4*)(sc + (long)e * 4) = make_float4(vv[0], vv[1], vv[2], vv[3]);
}

// ---------- exp + segment sum ----------
__global__ void edge_exp(const int* __restrict__ eidx, float* __restrict__ sc,
                         const unsigned* __restrict__ mx, float* __restrict__ den) {
    int e = blockIdx.x * blockDim.x + threadIdx.x;
    if (e >= N_EDGES) return;
    int dn = eidx[N_EDGES + e];
    float4 s4 = *(const float4*)(sc + (long)e * 4);
    uint4 m4 = *(const uint4*)(mx + (long)dn * 4);
    float p0 = __expf(s4.x - funflip(m4.x));
    float p1 = __expf(s4.y - funflip(m4.y));
    float p2 = __expf(s4.z - funflip(m4.z));
    float p3 = __expf(s4.w - funflip(m4.w));
    *(float4*)(sc + (long)e * 4) = make_float4(p0, p1, p2, p3);
    atomicAdd(&den[dn * 4 + 0], p0);
    atomicAdd(&den[dn * 4 + 1], p1);
    atomicAdd(&den[dn * 4 + 2], p2);
    atomicAdd(&den[dn * 4 + 3], p3);
}

// ---------- message pass: recompute e=ea@We+et via WMMA, scatter alpha*(hs+e) ----------
template <int OUT>
__global__ __launch_bounds__(256) void edge_message(const float* __restrict__ ea,
                                                    const int* __restrict__ eidx,
                                                    const int* __restrict__ etype,
                                                    const f16_t* __restrict__ Wp,
                                                    const float* __restrict__ ettab,
                                                    const float* __restrict__ p,
                                                    const float* __restrict__ den,
                                                    const float* __restrict__ hsrc,
                                                    float* __restrict__ agg) {
    int lane = threadIdx.x & 31, wid = threadIdx.x >> 5;
    int tile = blockIdx.x * (blockDim.x >> 5) + wid;
    int eb = tile * 16;
    if (eb >= N_EDGES) return;
    int erow = eb + (lane & 15);
    int base = (lane & 16) ? 8 : 0;
    v16h a;
    for (int g = 0; g < 2; g++) {
        int k0 = g * 16 + base;
        if (k0 < 16) cvt8(a, g, ea + (long)erow * 16 + k0);
        else         zero8(a, g);
    }
    int mh = (lane & 16) ? 8 : 0;
    int srcA[8], dstA[8], etA[8];
    float alphaA[8][4];
    for (int r = 0; r < 8; r++) {
        int em = eb + mh + r;
        srcA[r] = eidx[em]; dstA[r] = eidx[N_EDGES + em]; etA[r] = etype[em];
        float4 p4 = *(const float4*)(p + (long)em * 4);
        float4 d4 = *(const float4*)(den + (long)dstA[r] * 4);
        alphaA[r][0] = p4.x / (d4.x + 1e-16f);
        alphaA[r][1] = p4.y / (d4.y + 1e-16f);
        alphaA[r][2] = p4.z / (d4.z + 1e-16f);
        alphaA[r][3] = p4.w / (d4.w + 1e-16f);
    }
    const v16h* Bp = (const v16h*)Wp;
    constexpr int D = OUT / 4;
    for (int ct = 0; ct < OUT / 16; ct++) {
        int col = ct * 16 + (lane & 15);
        int hh = col / D;                 // head, uniform per (lane,ct)
        v16h b = Bp[ct * 32 + lane];
        v8f acc = {};
        acc = __builtin_amdgcn_wmma_f32_16x16x32_f16(false, a, false, b,
                                                     (short)0, acc, false, false);
        for (int r = 0; r < 8; r++) {
            float ev = acc[r] + ettab[etA[r] * OUT + col];
            float msg = alphaA[r][hh] * (hsrc[(long)srcA[r] * OUT + col] + ev);
            atomicAdd(&agg[(long)dstA[r] * OUT + col], msg);
        }
    }
}

// ---------- layer finalizers ----------
__global__ void finalize1(const float* __restrict__ agg1, const float* __restrict__ x,
                          const float* __restrict__ res, float* __restrict__ z1) {
    long idx = (long)blockIdx.x * blockDim.x + threadIdx.x;
    if (idx >= (long)N_NODES * 64) return;
    int n = (int)(idx >> 6), c = (int)(idx & 63);
    float v = agg1[idx];
    for (int j = 0; j < 16; j++) v += x[(long)n * 16 + j] * res[j * 64 + c];
    z1[idx] = v > 0.f ? v : 0.f;       // relu between layers
}

__global__ void finalize2(const float* __restrict__ agg2, const float* __restrict__ z1,
                          float* __restrict__ z) {
    long idx = (long)blockIdx.x * blockDim.x + threadIdx.x;
    if (idx >= (long)N_NODES * 64) return;
    int n = (int)(idx >> 6), c = (int)(idx & 63);
    const float* ap = agg2 + (long)n * 256;
    float v = ap[c] + ap[64 + c] + ap[128 + c] + ap[192 + c];
    z[idx] = z1[idx] + 0.25f * v;      // mean over heads + identity residual
}

// ---------- edge decoder: relu([z[r]||z[c]]@W1+b1)@W2+b2 via WMMA ----------
__global__ __launch_bounds__(256) void decoder(const float* __restrict__ z,
                                               const int* __restrict__ eli,
                                               const f16_t* __restrict__ W1p,
                                               const float* __restrict__ b1,
                                               const float* __restrict__ W2,
                                               const float* __restrict__ b2,
                                               float* __restrict__ pred) {
    int lane = threadIdx.x & 31, wid = threadIdx.x >> 5;
    int tile = blockIdx.x * (blockDim.x >> 5) + wid;
    int eb = tile * 16;
    if (eb >= N_EL) return;
    int em = eb + (lane & 15);
    int rI = eli[em], cI = eli[N_EL + em];
    int base = (lane & 16) ? 8 : 0;
    v16h a[4];
    for (int kk = 0; kk < 4; kk++)
        for (int g = 0; g < 2; g++) {
            int k0 = kk * 32 + g * 16 + base;
            const float* src = (k0 < 64) ? (z + (long)rI * 64 + k0)
                                         : (z + (long)cI * 64 + (k0 - 64));
            cvt8(a[kk], g, src);
        }
    const v16h* Bp = (const v16h*)W1p;
    int mh = (lane & 16) ? 8 : 0;
    float partial[8] = {0, 0, 0, 0, 0, 0, 0, 0};
    for (int ct = 0; ct < 4; ct++) {
        int col = ct * 16 + (lane & 15);
        v8f acc = {};
        for (int kk = 0; kk < 4; kk++) {
            v16h b = Bp[(kk * 4 + ct) * 32 + lane];
            acc = __builtin_amdgcn_wmma_f32_16x16x32_f16(false, a[kk], false, b,
                                                         (short)0, acc, false, false);
        }
        float bb = b1[col], w2 = W2[col];
        for (int r = 0; r < 8; r++) {
            float hv = acc[r] + bb;
            hv = hv > 0.f ? hv : 0.f;
            partial[r] += hv * w2;
        }
    }
    for (int m = 1; m < 16; m <<= 1)
        for (int r = 0; r < 8; r++) partial[r] += __shfl_xor(partial[r], m, 32);
    if ((lane & 15) == 0) {
        float bb2 = b2[0];
        for (int r = 0; r < 8; r++) pred[eb + mh + r] = partial[r] + bb2;
    }
}

// ---------- host ----------
extern "C" void kernel_launch(void* const* d_in, const int* in_sizes, int n_in,
                              void* d_out, int out_size, void* d_ws, size_t ws_size,
                              hipStream_t stream) {
    (void)in_sizes; (void)n_in; (void)out_size; (void)ws_size;
    const float* x    = (const float*)d_in[0];
    const int*   eidx = (const int*)d_in[1];
    const int*   ntyp = (const int*)d_in[2];
    const float* ea   = (const float*)d_in[3];
    const int*   etyp = (const int*)d_in[4];
    const int*   eli  = (const int*)d_in[5];
    const float* Wx1  = (const float*)d_in[6];
    const float* We1  = (const float*)d_in[7];
    const float* nt1  = (const float*)d_in[8];
    const float* et1  = (const float*)d_in[9];
    const float* as1  = (const float*)d_in[10];
    const float* ad1  = (const float*)d_in[11];
    const float* ae1  = (const float*)d_in[12];
    const float* res1 = (const float*)d_in[13];
    const float* Wx2  = (const float*)d_in[14];
    const float* We2  = (const float*)d_in[15];
    const float* nt2  = (const float*)d_in[16];
    const float* et2  = (const float*)d_in[17];
    const float* as2  = (const float*)d_in[18];
    const float* ad2  = (const float*)d_in[19];
    const float* ae2  = (const float*)d_in[20];
    const float* W1   = (const float*)d_in[21];
    const float* b1   = (const float*)d_in[22];
    const float* W2   = (const float*)d_in[23];
    const float* b2   = (const float*)d_in[24];

    float* ws = (float*)d_ws;
    size_t off = 0;
    float* h1   = ws + off; off += (size_t)N_NODES * 64;
    float* h2   = ws + off; off += (size_t)N_NODES * 256;
    float* z1   = ws + off; off += (size_t)N_NODES * 64;
    float* agg1 = ws + off; off += (size_t)N_NODES * 64;
    float* agg2 = ws + off; off += (size_t)N_NODES * 256;
    float* sc   = ws + off; off += (size_t)N_EDGES * 4;
    unsigned* mx = (unsigned*)(ws + off); off += (size_t)N_NODES * 4;
    float* den  = ws + off; off += (size_t)N_NODES * 4;
    float* ssrc = ws + off; off += (size_t)N_NODES * 4;
    float* sdst = ws + off; off += (size_t)N_NODES * 4;
    float* wea1 = ws + off; off += 64;
    float* eta1 = ws + off; off += 88;
    float* wea2 = ws + off; off += 64;
    float* eta2 = ws + off; off += 88;
    // packed B-fragment weights (f16), 32B-aligned (off is a multiple of 8 floats)
    f16_t* pWx1 = (f16_t*)(ws + off);
    f16_t* pWe1 = pWx1 + 1 * 4 * 512;     // (16,64):  KS=1, 4 tiles
    f16_t* pWx2 = pWe1 + 1 * 4 * 512;     // (16,64)
    f16_t* pWe2 = pWx2 + 2 * 16 * 512;    // (64,256): KS=2, 16 tiles
    f16_t* pW1  = pWe2 + 1 * 16 * 512;    // (16,256)
    //                  (128,64): KS=4, 4 tiles -> 8192 halves

    float* pred = (float*)d_out;
    float* z    = pred + N_EL;

    const int T = 256;
    // weight packing into WMMA B-fragment order
    pack_bfrag<16, 64>  <<<(2048  + T - 1) / T, T, 0, stream>>>(Wx1, pWx1);
    pack_bfrag<16, 64>  <<<(2048  + T - 1) / T, T, 0, stream>>>(We1, pWe1);
    pack_bfrag<64, 256> <<<(16384 + T - 1) / T, T, 0, stream>>>(Wx2, pWx2);
    pack_bfrag<16, 256> <<<(8192  + T - 1) / T, T, 0, stream>>>(We2, pWe2);
    pack_bfrag<128, 64> <<<(8192  + T - 1) / T, T, 0, stream>>>(W1, pW1);
    prep_edge_vecs<<<1, T, 0, stream>>>(We1, et1, ae1, wea1, eta1, 16);
    prep_edge_vecs<<<1, T, 0, stream>>>(We2, et2, ae2, wea2, eta2, 64);

    int nTileBlocks = (N_NODES / 16 + 7) / 8;           // 16 nodes/wave, 8 waves/block
    int eBlocks = (N_EDGES + T - 1) / T;
    int eTileBlocks = (N_EDGES / 16 + 7) / 8;
    int nodeHeadBlocks = (N_NODES * 4 + T - 1) / T;
    int ncBlocks = (N_NODES * 64 + T - 1) / T;

    // ---------- layer 1 ----------
    init_layer<<<2048, T, 0, stream>>>(agg1, (long)N_NODES * 64, mx, den, N_NODES * 4);
    node_proj<16, 64><<<nTileBlocks, T, 0, stream>>>(x, ntyp, nt1, pWx1, h1, N_NODES);
    score_dots<<<nodeHeadBlocks, T, 0, stream>>>(h1, as1, ad1, ssrc, sdst, N_NODES, 16);
    edge_score<<<eBlocks, T, 0, stream>>>(ea, eidx, etyp, ssrc, sdst, wea1, eta1, sc, mx);
    edge_exp<<<eBlocks, T, 0, stream>>>(eidx, sc, mx, den);
    edge_message<64><<<eTileBlocks, T, 0, stream>>>(ea, eidx, etyp, pWe1, et1, sc, den, h1, agg1);
    finalize1<<<ncBlocks, T, 0, stream>>>(agg1, x, res1, z1);

    // ---------- layer 2 ----------
    init_layer<<<2048, T, 0, stream>>>(agg2, (long)N_NODES * 256, mx, den, N_NODES * 4);
    node_proj<64, 256><<<nTileBlocks, T, 0, stream>>>(z1, ntyp, nt2, pWx2, h2, N_NODES);
    score_dots<<<nodeHeadBlocks, T, 0, stream>>>(h2, as2, ad2, ssrc, sdst, N_NODES, 64);
    edge_score<<<eBlocks, T, 0, stream>>>(ea, eidx, etyp, ssrc, sdst, wea2, eta2, sc, mx);
    edge_exp<<<eBlocks, T, 0, stream>>>(eidx, sc, mx, den);
    edge_message<256><<<eTileBlocks, T, 0, stream>>>(ea, eidx, etyp, pWe2, et2, sc, den, h2, agg2);
    finalize2<<<ncBlocks, T, 0, stream>>>(agg2, z1, z);

    // ---------- decoder ----------
    int dTileBlocks = (N_EL / 16 + 7) / 8;
    decoder<<<dTileBlocks, T, 0, stream>>>(z, eli, pW1, b1, W2, b2, pred);
}
